// Model_13134009991755
// MI455X (gfx1250) — matrix operand order repaired and checked by reading, not compiled, not run on hardware
//
#include <hip/hip_runtime.h>
#include <hip/hip_bf16.h>

// ---------------------------------------------------------------------------
// VMamba SS2D block for MI455X (gfx1250, wave32).
//   1) in_proj  : xz(8192,384) = x(8192,96) @ in_proj_w^T           [WMMA]
//   2) conv+SiLU: xc(B,L,192)  = silu(dwconv3x3(xz[:,:192]) + b)
//   3) x_proj   : xdbl(B,K,L,48) = xs @ x_proj_w^T (dir-gathered)   [WMMA]
//   4) scan     : 32 blocks x 192 thr; state[16] in VGPRs;
//                 per-step (dt,B,C) staged in LDS via TDM double-buffer
//   5) LN+gate  : shfl/LDS reduction over D=192, silu(z) gate
//   6) out_proj : out(8192,96) = yg(8192,192) @ out_proj_w^T        [WMMA]
// ---------------------------------------------------------------------------

typedef __attribute__((ext_vector_type(16))) _Float16 v16h;
typedef __attribute__((ext_vector_type(8)))  float    v8f;
typedef unsigned int v4u __attribute__((ext_vector_type(4)));
typedef int          v4i __attribute__((ext_vector_type(4)));
typedef int          v8i __attribute__((ext_vector_type(8)));

#define D_MODEL 96
#define D_INNER 192
#define D_STATE 16
#define DT_RANK 6
#define KDIR    4
#define BATCH   8
#define LSEQ    1024          // 32x32
#define ROWS    (BATCH*LSEQ)  // 8192
#define XDBL_C  48            // 38 padded to 48
#define CHUNK   64            // scan steps staged per TDM transfer

// direction bijection: sequence step l -> spatial row-major index
__device__ __forceinline__ int dir_idx(int k, int l) {
    int ll = (k & 2) ? (LSEQ - 1 - l) : l;              // reversed scans
    return (k & 1) ? (((ll & 31) << 5) | (ll >> 5))     // transpose
                   : ll;
}

__device__ __forceinline__ void pack8(v16h& v, float4 q0, float4 q1, int base) {
    v[base + 0] = (_Float16)q0.x; v[base + 1] = (_Float16)q0.y;
    v[base + 2] = (_Float16)q0.z; v[base + 3] = (_Float16)q0.w;
    v[base + 4] = (_Float16)q1.x; v[base + 5] = (_Float16)q1.y;
    v[base + 6] = (_Float16)q1.z; v[base + 7] = (_Float16)q1.w;
}

// A fragment 16x32 f16: lane-half g holds K = {8g..8g+7, 16+8g..23+8g}
__device__ __forceinline__ v16h load_a_frag(const float* arow, int k0, int g) {
    float4 q0 = *(const float4*)(arow + k0 + 8 * g);
    float4 q1 = *(const float4*)(arow + k0 + 8 * g + 4);
    float4 q2 = *(const float4*)(arow + k0 + 8 * g + 16);
    float4 q3 = *(const float4*)(arow + k0 + 8 * g + 20);
    v16h a;
    pack8(a, q0, q1, 0);
    pack8(a, q2, q3, 8);
    return a;
}

// B fragment 32x16 f16: lane-half g holds K = {8g..8g+7, 16+8g..23+8g}, N = lane&15
__device__ __forceinline__ v16h load_b_frag(const float* wrow, int k0, int g) {
    float4 q0 = *(const float4*)(wrow + k0 + 8 * g);
    float4 q1 = *(const float4*)(wrow + k0 + 8 * g + 4);
    float4 q2 = *(const float4*)(wrow + k0 + 8 * g + 16);
    float4 q3 = *(const float4*)(wrow + k0 + 8 * g + 20);
    v16h b;
    pack8(b, q0, q1, 0);
    pack8(b, q2, q3, 8);
    return b;
}

// TDM: contiguous 1-D copy of nelem dwords global->LDS (ISA 08_async_tensor §8).
__device__ __forceinline__ void tdm_load_1d(void* lds_ptr, const void* gptr, int nelem) {
    unsigned lds = (unsigned)(uintptr_t)lds_ptr;  // low 32 bits = LDS byte offset
    unsigned long long ga = (unsigned long long)(uintptr_t)gptr;
    v4u g0;
    g0.x = 1u;                                       // count=1 user descriptor
    g0.y = lds;                                      // lds_addr
    g0.z = (unsigned)(ga & 0xFFFFFFFFu);             // global_addr[31:0]
    g0.w = (unsigned)((ga >> 32) & 0x1FFFFFFu) | (2u << 30);  // addr[56:32] | type=2
    v8i g1;
    g1[0] = 0x00020000;                              // data_size=2 (4B), mask=0
    g1[1] = (int)((unsigned)(nelem & 0xFFFF) << 16); // tensor_dim0[15:0]
    g1[2] = (int)(((unsigned)nelem >> 16) | (1u << 16)); // tensor_dim0[31:16] | tensor_dim1=1
    g1[3] = (int)((unsigned)(nelem & 0xFFFF) << 16); // tile_dim0 = nelem
    g1[4] = 0;                                       // tile_dim1/2 unused
    g1[5] = nelem;                                   // tensor_dim0_stride[31:0]
    g1[6] = 0;
    g1[7] = 0;
    v4i z4 = {0, 0, 0, 0};
#if defined(__clang_major__) && (__clang_major__ >= 23)
    v8i z8 = {0, 0, 0, 0, 0, 0, 0, 0};
    __builtin_amdgcn_tensor_load_to_lds(g0, g1, z4, z4, z8, 0);
#else
    __builtin_amdgcn_tensor_load_to_lds(g0, g1, z4, z4, 0);
#endif
}

// ---------------------------------------------------------------------------
// C(MxN) = A(MxKD) * W(NxKD)^T. 4 waves/block, one 16x16 tile per wave (M).
// Out-of-range output columns read a clamped (valid) weight row; their
// results are discarded at the store, so no predication around the WMMA.
// ---------------------------------------------------------------------------
template <int KD, int N>
__global__ void wmma_gemm_nt(const float* __restrict__ A, int lda,
                             const float* __restrict__ W, int ldw,
                             float* __restrict__ C, int ldc) {
    const int lane = threadIdx.x & 31;
    const int wv   = threadIdx.x >> 5;             // 0..3
    const int g    = lane >> 4;
    const int mt   = blockIdx.y * 4 + wv;
    const int mrow = mt * 16 + (lane & 15);
    int ncol = blockIdx.x * 16 + (lane & 15);
    const bool nvalid = (ncol < N);
    if (!nvalid) ncol = N - 1;                     // clamp: stay in-bounds
    const float* arow = A + (size_t)mrow * lda;
    const float* wrow = W + (size_t)ncol * ldw;

    v8f acc = {};
#pragma unroll
    for (int k0 = 0; k0 < KD; k0 += 32) {
        v16h a = load_a_frag(arow, k0, g);
        v16h b = load_b_frag(wrow, k0, g);
        acc = __builtin_amdgcn_wmma_f32_16x16x32_f16(false, a, false, b,
                                                     (short)0, acc, false, false);
    }
#pragma unroll
    for (int r = 0; r < 8; ++r) {
        int row = mt * 16 + r + 8 * g;             // C/D: M = r + 8*lane_half
        int col = blockIdx.x * 16 + (lane & 15);
        if (col < N) C[(size_t)row * ldc + col] = acc[r];
    }
}

// ---------------------------------------------------------------------------
// Depthwise 3x3 conv (SAME) + bias + SiLU on xz[:, :192] -> xc(B,L,192)
// ---------------------------------------------------------------------------
__global__ void conv_silu(const float* __restrict__ xz,
                          const float* __restrict__ cw,
                          const float* __restrict__ cb,
                          float* __restrict__ xc) {
    int t = blockIdx.x * blockDim.x + threadIdx.x;
    if (t >= ROWS * D_INNER) return;
    int d   = t % D_INNER;
    int pos = t / D_INNER;
    int b   = pos >> 10;
    int hw  = pos & 1023;
    int hh  = hw >> 5, ww = hw & 31;
    float acc = cb[d];
#pragma unroll
    for (int dy = 0; dy < 3; ++dy) {
        int y = hh + dy - 1;
        if (y < 0 || y > 31) continue;
#pragma unroll
        for (int dx = 0; dx < 3; ++dx) {
            int x = ww + dx - 1;
            if (x < 0 || x > 31) continue;
            acc += cw[d * 9 + dy * 3 + dx] *
                   xz[(((size_t)(b << 10) + (y << 5) + x) * (2 * D_INNER)) + d];
        }
    }
    xc[(size_t)pos * D_INNER + d] = acc / (1.f + __expf(-acc));
}

// ---------------------------------------------------------------------------
// x_proj: xdbl(b,k,l,0:38) = x_proj_w[k](38x192) @ xs[b,k,:,l].
// A rows gathered through the direction bijection; N padded 38->48.
// ---------------------------------------------------------------------------
__global__ void xproj_wmma(const float* __restrict__ xc,
                           const float* __restrict__ xpw,
                           float* __restrict__ xdbl) {
    const int k    = blockIdx.z;
    const int lane = threadIdx.x & 31;
    const int wv   = threadIdx.x >> 5;
    const int g    = lane >> 4;
    const int mt   = blockIdx.y * 4 + wv;
    const int m    = mt * 16 + (lane & 15);        // (b,l)
    const int b    = m >> 10, l = m & 1023;
    const float* arow = xc + ((size_t)(b << 10) + dir_idx(k, l)) * D_INNER;
    int n = blockIdx.x * 16 + (lane & 15);
    const bool nvalid = (n < DT_RANK + 2 * D_STATE);
    if (!nvalid) n = DT_RANK + 2 * D_STATE - 1;    // clamp
    const float* wrow = xpw + (size_t)k * (DT_RANK + 2 * D_STATE) * D_INNER
                            + (size_t)n * D_INNER;

    v8f acc = {};
#pragma unroll
    for (int k0 = 0; k0 < D_INNER; k0 += 32) {
        v16h a = load_a_frag(arow, k0, g);
        v16h bb = load_b_frag(wrow, k0, g);
        acc = __builtin_amdgcn_wmma_f32_16x16x32_f16(false, a, false, bb,
                                                     (short)0, acc, false, false);
    }
#pragma unroll
    for (int r = 0; r < 8; ++r) {
        int rowm = mt * 16 + r + 8 * g;
        int col  = blockIdx.x * 16 + (lane & 15);
        if (col < DT_RANK + 2 * D_STATE) {
            int b2 = rowm >> 10, l2 = rowm & 1023;
            xdbl[((size_t)(b2 * KDIR + k) * LSEQ + l2) * XDBL_C + col] = acc[r];
        }
    }
}

// ---------------------------------------------------------------------------
// Selective scan. One block per (b,k) = 32 blocks x 192 threads.
// Thread d keeps h[16] in VGPRs. The per-step (dt_raw,B,C) stream is staged
// in 64-step chunks into double-buffered LDS by the Tensor Data Mover
// (TENSOR_LOAD_TO_LDS issued by wave 0, s_wait_tensorcnt + barrier per chunk).
// ---------------------------------------------------------------------------
__global__ void scan_kernel(const float* __restrict__ xdbl,
                            const float* __restrict__ xc,
                            const float* __restrict__ dtw,
                            const float* __restrict__ dtb,
                            const float* __restrict__ Alog,
                            const float* __restrict__ Dsv,
                            float* __restrict__ ys) {
    __shared__ float sbuf[2][CHUNK * XDBL_C];      // 2 x 12 KB
    const int b  = blockIdx.x >> 2;
    const int k  = blockIdx.x & 3;
    const int d  = threadIdx.x;                    // 0..191
    const int kd = k * D_INNER + d;

    float Arow[D_STATE];
#pragma unroll
    for (int n = 0; n < D_STATE; ++n)
        Arow[n] = -__expf(Alog[(size_t)kd * D_STATE + n]);
    float w6[DT_RANK];
#pragma unroll
    for (int r = 0; r < DT_RANK; ++r) w6[r] = dtw[(size_t)kd * DT_RANK + r];
    const float bias = dtb[kd];
    const float Dd   = Dsv[kd];

    float h[D_STATE];
#pragma unroll
    for (int n = 0; n < D_STATE; ++n) h[n] = 0.f;

    const float* xd = xdbl + (size_t)((b * KDIR + k) << 10) * XDBL_C;

    // prologue: stage chunk 0
    if (threadIdx.x < 32) {
        tdm_load_1d(&sbuf[0][0], xd, CHUNK * XDBL_C);
        __builtin_amdgcn_s_wait_tensorcnt(0);
    }
    __syncthreads();

    for (int c = 0; c < LSEQ / CHUNK; ++c) {
        const int cur = c & 1;
        if ((c + 1) < LSEQ / CHUNK && threadIdx.x < 32)
            tdm_load_1d(&sbuf[(c + 1) & 1][0],
                        xd + (size_t)(c + 1) * CHUNK * XDBL_C, CHUNK * XDBL_C);

        for (int s = 0; s < CHUNK; ++s) {
            const int l = c * CHUNK + s;
            const float* sm = &sbuf[cur][s * XDBL_C];
            const int idx = dir_idx(k, l);
            const float xv = xc[((size_t)(b << 10) + idx) * D_INNER + d];
            if (l + 1 < LSEQ)
                __builtin_prefetch(xc + ((size_t)(b << 10) + dir_idx(k, l + 1)) * D_INNER + d,
                                   0, 0);          // global_prefetch_b8

            float dt = bias;
#pragma unroll
            for (int r = 0; r < DT_RANK; ++r) dt += sm[r] * w6[r];
            const float delta = (dt > 15.f) ? dt : __logf(1.f + __expf(dt));

            float y = Dd * xv;
#pragma unroll
            for (int n = 0; n < D_STATE; ++n) {
                float dA = __expf(delta * Arow[n]);
                h[n] = dA * h[n] + (delta * sm[DT_RANK + n]) * xv;
                y += h[n] * sm[DT_RANK + D_STATE + n];
            }
            ys[((size_t)((b * KDIR + k) << 10) + idx) * D_INNER + d] = y;
        }
        if (threadIdx.x < 32) __builtin_amdgcn_s_wait_tensorcnt(0);
        __syncthreads();
    }
}

// ---------------------------------------------------------------------------
// Combine 4 directions + LayerNorm(D=192) + SiLU(z) gate -> yg(B,L,192).
// ---------------------------------------------------------------------------
__global__ void ln_gate(const float* __restrict__ ys,
                        const float* __restrict__ xz,
                        const float* __restrict__ gw,
                        const float* __restrict__ gb,
                        float* __restrict__ yg) {
    __shared__ float ps[6], ps2[6];
    const int pos = blockIdx.x;                    // (b,idx)
    const int b = pos >> 10, idx = pos & 1023;
    const int d = threadIdx.x;

    float y = 0.f;
#pragma unroll
    for (int k = 0; k < KDIR; ++k)
        y += ys[((size_t)((b * KDIR + k) << 10) + idx) * D_INNER + d];

    float s = y, s2 = y * y;
#pragma unroll
    for (int m = 16; m >= 1; m >>= 1) {
        s  += __shfl_xor(s,  m, 32);
        s2 += __shfl_xor(s2, m, 32);
    }
    if ((d & 31) == 0) { ps[d >> 5] = s; ps2[d >> 5] = s2; }
    __syncthreads();
    if (d == 0) {
        float t = 0.f, t2 = 0.f;
#pragma unroll
        for (int i = 0; i < 6; ++i) { t += ps[i]; t2 += ps2[i]; }
        ps[0]  = t  * (1.f / D_INNER);
        ps2[0] = t2 * (1.f / D_INNER);
    }
    __syncthreads();
    const float mu  = ps[0];
    const float var = ps2[0] - mu * mu;
    const float yn  = (y - mu) * rsqrtf(var + 1e-5f) * gw[d] + gb[d];
    const float z   = xz[(size_t)pos * (2 * D_INNER) + D_INNER + d];
    const float sg  = 1.f / (1.f + __expf(-z));
    yg[(size_t)pos * D_INNER + d] = yn * (z * sg);
}

// ---------------------------------------------------------------------------
extern "C" void kernel_launch(void* const* d_in, const int* in_sizes, int n_in,
                              void* d_out, int out_size, void* d_ws, size_t ws_size,
                              hipStream_t stream) {
    const float* x    = (const float*)d_in[0];   // (8,32,32,96)
    const float* inw  = (const float*)d_in[1];   // (384,96)
    const float* cw   = (const float*)d_in[2];   // (192,1,3,3)
    const float* cb   = (const float*)d_in[3];   // (192,)
    const float* xpw  = (const float*)d_in[4];   // (4,38,192)
    const float* dtw  = (const float*)d_in[5];   // (4,192,6)
    const float* dtb  = (const float*)d_in[6];   // (4,192)
    const float* Alog = (const float*)d_in[7];   // (768,16)
    const float* Dsv  = (const float*)d_in[8];   // (768,)
    const float* gw   = (const float*)d_in[9];   // (192,)
    const float* gb   = (const float*)d_in[10];  // (192,)
    const float* outw = (const float*)d_in[11];  // (96,192)
    float* out = (float*)d_out;                  // (8,32,32,96)

    float* ws   = (float*)d_ws;
    float* xz   = ws;                                    // 8192*384
    float* xc   = xz   + (size_t)ROWS * 2 * D_INNER;     // 8192*192
    float* xdbl = xc   + (size_t)ROWS * D_INNER;         // 8192*4*48
    float* ys   = xdbl + (size_t)ROWS * KDIR * XDBL_C;   // 8192*4*192
    float* yg   = ys   + (size_t)ROWS * KDIR * D_INNER;  // 8192*192

    // 1) in_proj: (8192x96)x(96x384); 4 M-tiles per block
    wmma_gemm_nt<D_MODEL, 2 * D_INNER>
        <<<dim3((2 * D_INNER) / 16, ROWS / 64), dim3(128), 0, stream>>>(
            x, D_MODEL, inw, D_MODEL, xz, 2 * D_INNER);

    // 2) depthwise conv + SiLU
    conv_silu<<<dim3((ROWS * D_INNER + 255) / 256), dim3(256), 0, stream>>>(
        xz, cw, cb, xc);

    // 3) x_proj per direction (N=38 padded to 48)
    xproj_wmma<<<dim3(XDBL_C / 16, ROWS / 64, KDIR), dim3(128), 0, stream>>>(
        xc, xpw, xdbl);

    // 4) selective scan (TDM-staged)
    scan_kernel<<<dim3(BATCH * KDIR), dim3(D_INNER), 0, stream>>>(
        xdbl, xc, dtw, dtb, Alog, Dsv, ys);

    // 5) LayerNorm + gate
    ln_gate<<<dim3(ROWS), dim3(D_INNER), 0, stream>>>(ys, xz, gw, gb, yg);

    // 6) out_proj: (8192x192)x(192x96)
    wmma_gemm_nt<D_INNER, D_MODEL>
        <<<dim3(D_MODEL / 16, ROWS / 64), dim3(128), 0, stream>>>(
            yg, D_INNER, outw, D_INNER, out, D_MODEL);
}
